// FastInvariantPointAttention_36575941493400
// MI455X (gfx1250) — matrix operand order repaired and checked by reading, not compiled, not run on hardware
//
#include <hip/hip_runtime.h>
#include <hip/hip_bf16.h>
#include <math.h>

// Problem constants (B=1)
#define S_   1024
#define CS_  384
#define CZ_  128
#define CH_  256
#define H_   8
#define PV_  8
#define HD_  32
#define INF_ 100000.0f

typedef __attribute__((ext_vector_type(16))) _Float16 v16h;
typedef __attribute__((ext_vector_type(8)))  _Float16 v8h;
typedef __attribute__((ext_vector_type(8)))  float    v8f;
typedef int v4i_ __attribute__((vector_size(16)));

#define GLOBAL_AS __attribute__((address_space(1)))
#define LDS_AS    __attribute__((address_space(3)))

#if __has_builtin(__builtin_amdgcn_global_load_async_to_lds_b128)
#define HAVE_ASYNC_LDS 1
#else
#define HAVE_ASYNC_LDS 0
#endif

__device__ __forceinline__ void wait_async0() {
#if HAVE_ASYNC_LDS
#if __has_builtin(__builtin_amdgcn_s_wait_asynccnt)
  __builtin_amdgcn_s_wait_asynccnt(0);
#else
  asm volatile("s_wait_asynccnt 0x0" ::: "memory");
#endif
#endif
}

// ---------------------------------------------------------------------------
// WMMA fragment layouts (wave32, 16x16x32 f16; ISA 7.12.2):
// A (16x32): lane m=lane&15 holds row m; this code's order: element e -> K = 16*hi + e,
//            hi = lane>>4.
//            => one contiguous 16-half run per lane (two 16B loads).
// B (32x16): lane holds column n=lane&15; this code's order: element e -> K = 16*hi + e.
//            => contiguous 16 halves per lane IF operand is stored K-major
//               per column (i.e. transposed tile).
// C (16x16 f32): vgpr r -> row = r + 8*hi, col = lane&15.
// ---------------------------------------------------------------------------
__device__ __forceinline__ v16h load_a_frag(const _Float16* row, int lane) {
  int hi = (lane >> 4) & 1;
  v8h lo = *(const v8h*)(row + 16 * hi);
  v8h hp = *(const v8h*)(row + 16 * hi + 8);
  v16h r;
  #pragma unroll
  for (int e = 0; e < 8; ++e) { r[e] = lo[e]; r[8 + e] = hp[e]; }
  return r;
}

// col points at &T[n][k0] where T is stored [n][K] (K-contiguous per column n)
__device__ __forceinline__ v16h load_b_frag(const _Float16* col, int lane) {
  int hi = (lane >> 4) & 1;
  const _Float16* p = col + 16 * hi;
  v8h lo = *(const v8h*)(p);
  v8h hp = *(const v8h*)(p + 8);
  v16h r;
  #pragma unroll
  for (int e = 0; e < 8; ++e) { r[e] = lo[e]; r[8 + e] = hp[e]; }
  return r;
}

// ---------------------------------------------------------------------------
// Kernel 1: LayerNorm of s -> s_n (f16).  One wave per row.
// ---------------------------------------------------------------------------
__global__ void k_ln_s(const float* __restrict__ s,
                       const float* __restrict__ g,
                       const float* __restrict__ b,
                       _Float16* __restrict__ s_n) {
  int row  = blockIdx.x * (blockDim.x >> 5) + (threadIdx.x >> 5);
  int lane = threadIdx.x & 31;
  if (row >= S_) return;
  const float* x = s + (size_t)row * CS_;
  float sum = 0.f, sq = 0.f;
  for (int c = lane; c < CS_; c += 32) { float v = x[c]; sum += v; sq += v * v; }
  #pragma unroll
  for (int m = 16; m >= 1; m >>= 1) {
    sum += __shfl_xor(sum, m, 32);
    sq  += __shfl_xor(sq,  m, 32);
  }
  float mean = sum * (1.0f / CS_);
  float var  = sq * (1.0f / CS_) - mean * mean;
  float inv  = rsqrtf(var + 1e-5f);
  for (int c = lane; c < CS_; c += 32) {
    float v = (x[c] - mean) * inv * g[c] + b[c];
    s_n[(size_t)row * CS_ + c] = (_Float16)v;
  }
}

// ---------------------------------------------------------------------------
// Kernel 2: projection GEMM  out[i,n] = sum_k s_n[i,k] * W[n,k] (+bias)
// One wave per 16x16 output tile, K=384 in 12 WMMA steps.
// ---------------------------------------------------------------------------
template <bool F16OUT>
__global__ void k_proj(const _Float16* __restrict__ sn,
                       const float* __restrict__ W,
                       const float* __restrict__ bias,
                       void* __restrict__ out, int NCOL) {
  int tiles_n = NCOL >> 4;
  int waveId  = blockIdx.x * (blockDim.x >> 5) + (threadIdx.x >> 5);
  int lane    = threadIdx.x & 31;
  int ti = waveId / tiles_n, tn = waveId % tiles_n;
  if (ti >= (S_ >> 4)) return;
  int i0 = ti << 4, n0 = tn << 4;
  int M = i0 + (lane & 15);
  int N = n0 + (lane & 15);
  int hi = (lane >> 4) & 1;
  v8f acc = {};
  for (int k0 = 0; k0 < CS_; k0 += 32) {
    v16h a = load_a_frag(sn + (size_t)M * CS_ + k0, lane);
    const float* wp = W + (size_t)N * CS_ + k0 + 16 * hi; // 16 contiguous f32
    v16h bm;
    #pragma unroll
    for (int e = 0; e < 16; ++e) bm[e] = (_Float16)wp[e];
    acc = __builtin_amdgcn_wmma_f32_16x16x32_f16(false, a, false, bm,
                                                 (short)0, acc, false, false);
  }
  #pragma unroll
  for (int r = 0; r < 8; ++r) {
    int i = i0 + r + 8 * hi;
    int n = n0 + (lane & 15);
    float v = acc[r] + (bias ? bias[n] : 0.0f);
    if (F16OUT) ((_Float16*)out)[(size_t)i * NCOL + n] = (_Float16)v;
    else        ((float*)out)[(size_t)i * NCOL + n] = v;
  }
}

// ---------------------------------------------------------------------------
// Kernel 3: coords + v_points
// ---------------------------------------------------------------------------
__global__ void k_vpoints(const float* __restrict__ rigids,
                          const float* __restrict__ pt,
                          float* __restrict__ coords,
                          float* __restrict__ vpts) {
  int j = blockIdx.x * blockDim.x + threadIdx.x;
  if (j >= S_) return;
  float cx = rigids[(size_t)j * 16 + 0 * 4 + 3];
  float cy = rigids[(size_t)j * 16 + 1 * 4 + 3];
  float cz = rigids[(size_t)j * 16 + 2 * 4 + 3];
  coords[j * 3 + 0] = cx; coords[j * 3 + 1] = cy; coords[j * 3 + 2] = cz;
  for (int h = 0; h < H_; ++h) {
    float sum = 0.f;
    #pragma unroll
    for (int p = 0; p < PV_; ++p) sum += pt[(size_t)j * (H_ * PV_) + h * PV_ + p];
    vpts[(size_t)j * 24 + h * 3 + 0] = sum * cx;
    vpts[(size_t)j * 24 + h * 3 + 1] = sum * cy;
    vpts[(size_t)j * 24 + h * 3 + 2] = sum * cz;
  }
}

// ---------------------------------------------------------------------------
// Kernel 4: pair bias.  Streams z (537 MB - the bandwidth-dominant pass):
// one coalesced float4 per lane per pair, LayerNorm in registers, f16 tile
// in LDS, WMMA (16 pairs x K=128 x heads) against W_pair^T stored K-major.
// ---------------------------------------------------------------------------
__global__ void k_pairbias(const float* __restrict__ z,
                           const float* __restrict__ gz,
                           const float* __restrict__ bz,
                           const float* __restrict__ Wp,
                           float* __restrict__ bias) {
  __shared__ _Float16 Az[4][16][CZ_];   // per-wave normalized z tile [pair][k]
  __shared__ _Float16 Bp[16][CZ_];      // W_pair, K-major per column n (n>=8 -> 0)
  int w    = threadIdx.x >> 5;
  int lane = threadIdx.x & 31;
  int t    = threadIdx.x;
  // Fill Bp transposed: column n gets W_pair[n][*] contiguous in K
  #pragma unroll
  for (int n = 0; n < 16; ++n)
    Bp[n][t] = (n < H_) ? (_Float16)Wp[(size_t)n * CZ_ + t] : (_Float16)0.0f;
  __syncthreads();

  size_t p0 = ((size_t)blockIdx.x * 4 + w) * 16;
  float g0 = gz[lane * 4 + 0], g1 = gz[lane * 4 + 1], g2 = gz[lane * 4 + 2], g3 = gz[lane * 4 + 3];
  float b0 = bz[lane * 4 + 0], b1 = bz[lane * 4 + 1], b2 = bz[lane * 4 + 2], b3 = bz[lane * 4 + 3];

  for (int q = 0; q < 16; ++q) {
    size_t pair = p0 + q;
    const float4* zp = (const float4*)(z + pair * CZ_);
    float4 v = zp[lane];
    float sum = v.x + v.y + v.z + v.w;
    float sq  = v.x * v.x + v.y * v.y + v.z * v.z + v.w * v.w;
    #pragma unroll
    for (int m = 16; m >= 1; m >>= 1) {
      sum += __shfl_xor(sum, m, 32);
      sq  += __shfl_xor(sq,  m, 32);
    }
    float mean = sum * (1.0f / CZ_);
    float var  = sq * (1.0f / CZ_) - mean * mean;
    float inv  = rsqrtf(var + 1e-5f);
    _Float16 h0 = (_Float16)((v.x - mean) * inv * g0 + b0);
    _Float16 h1 = (_Float16)((v.y - mean) * inv * g1 + b1);
    _Float16 h2 = (_Float16)((v.z - mean) * inv * g2 + b2);
    _Float16 h3 = (_Float16)((v.w - mean) * inv * g3 + b3);
    *(ushort4*)&Az[w][q][lane * 4] =
        make_ushort4(*(unsigned short*)&h0, *(unsigned short*)&h1,
                     *(unsigned short*)&h2, *(unsigned short*)&h3);
  }
  __syncthreads();   // uniform: order LDS writes vs WMMA fragment reads

  v8f acc = {};
  #pragma unroll
  for (int k0 = 0; k0 < CZ_; k0 += 32) {
    v16h a  = load_a_frag(&Az[w][lane & 15][k0], lane);
    v16h bm = load_b_frag(&Bp[lane & 15][k0], lane);
    acc = __builtin_amdgcn_wmma_f32_16x16x32_f16(false, a, false, bm,
                                                 (short)0, acc, false, false);
  }
  int h = lane & 15;
  if (h < H_) {
    #pragma unroll
    for (int r = 0; r < 8; ++r) {
      size_t pair = p0 + r + 8 * ((lane >> 4) & 1);
      int i = (int)(pair >> 10), j = (int)(pair & 1023);
      bias[(((size_t)h * S_) + i) * S_ + j] = acc[r];
    }
  }
}

// ---------------------------------------------------------------------------
// Kernel 5: flash attention.  Block = 4 waves; each wave owns one 16-row
// query tile of one head.  j-loop step 32: 2 WMMA for Q.K^T, online softmax,
// 2 WMMA for P.V.  K tile loaded async-to-LDS; V stored transposed so P.V
// B-frags are contiguous.  Output ao in f16 (consumed only by final GEMM).
// ---------------------------------------------------------------------------
__global__ void k_attn(const _Float16* __restrict__ q,
                       const _Float16* __restrict__ k,
                       const _Float16* __restrict__ v,
                       const float* __restrict__ bias,
                       const float* __restrict__ mask,
                       _Float16* __restrict__ ao) {
  __shared__ _Float16 Kt[32][32];       // [j][d]  (B-frag for Q.K^T: row n=j, K=d)
  __shared__ _Float16 Vtt[32][32];      // [d][j]  (B-frag for P.V:  row n=d, K=j)
  __shared__ _Float16 Ps[4][16][32];    // per-wave P tile [row][j]
  int w    = threadIdx.x >> 5;
  int lane = threadIdx.x & 31;
  int h    = blockIdx.x >> 4;               // H_ * (S_/16/4) = 128 blocks
  int qt   = ((blockIdx.x & 15) << 2) + w;
  int i0   = qt << 4;
  int hi   = (lane >> 4) & 1;
  const float scale = 0.17677669529663687f; // 1/sqrt(32)

  v16h qa = load_a_frag(q + (size_t)(i0 + (lane & 15)) * CH_ + h * HD_, lane);

  v8f acc0 = {}, acc1 = {};
  float m_r[8], l_r[8];
  #pragma unroll
  for (int r = 0; r < 8; ++r) { m_r[r] = -1e30f; l_r[r] = 0.f; }

  for (int j0 = 0; j0 < S_; j0 += 32) {
    __syncthreads();
    {
      int row = threadIdx.x >> 2, c8 = (threadIdx.x & 3) * 8; // 128 thr x 8 halves
      const _Float16* kg = &k[(size_t)(j0 + row) * CH_ + h * HD_ + c8];
#if HAVE_ASYNC_LDS
      __builtin_amdgcn_global_load_async_to_lds_b128(
          (GLOBAL_AS v4i_*)kg,
          (LDS_AS v4i_*)&Kt[row][c8], 0, 0);
#else
      *(float4*)&Kt[row][c8] = *(const float4*)kg;
#endif
      // V: load 8 halves of row j, scatter transposed into Vtt[d][j]
      v8h vv = *(const v8h*)&v[(size_t)(j0 + row) * CH_ + h * HD_ + c8];
      #pragma unroll
      for (int u = 0; u < 8; ++u) Vtt[c8 + u][row] = vv[u];
    }
    wait_async0();
    __syncthreads();

    // S = Q (16x32) @ K^T (32x16): B-frag columns are K rows (contiguous d)
    v16h b1 = load_b_frag(&Kt[(lane & 15)][0], lane);
    v16h b2 = load_b_frag(&Kt[16 + (lane & 15)][0], lane);
    v8f zc = {};
    v8f s1 = __builtin_amdgcn_wmma_f32_16x16x32_f16(false, qa, false, b1, (short)0, zc, false, false);
    v8f s2 = __builtin_amdgcn_wmma_f32_16x16x32_f16(false, qa, false, b2, (short)0, zc, false, false);

    // online softmax over this 16x32 stripe
    #pragma unroll
    for (int r = 0; r < 8; ++r) {
      int i  = i0 + r + 8 * hi;
      int j1 = j0 + (lane & 15), j2 = j1 + 16;
      float x1 = s1[r] * scale + bias[(((size_t)h * S_) + i) * S_ + j1] + INF_ * (mask[j1] - 1.f);
      float x2 = s2[r] * scale + bias[(((size_t)h * S_) + i) * S_ + j2] + INF_ * (mask[j2] - 1.f);
      float mx = fmaxf(x1, x2);
      #pragma unroll
      for (int mm = 8; mm >= 1; mm >>= 1) mx = fmaxf(mx, __shfl_xor(mx, mm, 32));
      float mnew  = fmaxf(m_r[r], mx);
      float alpha = __expf(m_r[r] - mnew);
      float p1 = __expf(x1 - mnew), p2 = __expf(x2 - mnew);
      float ps = p1 + p2;
      #pragma unroll
      for (int mm = 8; mm >= 1; mm >>= 1) ps += __shfl_xor(ps, mm, 32);
      l_r[r] = l_r[r] * alpha + ps;
      m_r[r] = mnew;
      acc0[r] *= alpha;
      acc1[r] *= alpha;
      int rr = r + 8 * hi;
      Ps[w][rr][(lane & 15)]      = (_Float16)p1;
      Ps[w][rr][16 + (lane & 15)] = (_Float16)p2;
    }
    // Ps is per-wave; DS ops of one wave are in-order -> wave barrier suffices
    __builtin_amdgcn_wave_barrier();

    v16h pa  = load_a_frag(&Ps[w][(lane & 15)][0], lane);
    v16h bv0 = load_b_frag(&Vtt[(lane & 15)][0], lane);
    v16h bv1 = load_b_frag(&Vtt[16 + (lane & 15)][0], lane);
    acc0 = __builtin_amdgcn_wmma_f32_16x16x32_f16(false, pa, false, bv0, (short)0, acc0, false, false);
    acc1 = __builtin_amdgcn_wmma_f32_16x16x32_f16(false, pa, false, bv1, (short)0, acc1, false, false);
  }

  #pragma unroll
  for (int r = 0; r < 8; ++r) {
    int i = i0 + r + 8 * hi;
    float invl = 1.0f / (l_r[r] + 1e-30f);
    ao[(size_t)i * CH_ + h * HD_ + (lane & 15)]      = (_Float16)(acc0[r] * invl);
    ao[(size_t)i * CH_ + h * HD_ + 16 + (lane & 15)] = (_Float16)(acc1[r] * invl);
  }
}

// ---------------------------------------------------------------------------
// Kernel 6: distance-weighted point aggregation.  One wave per output row i.
// Output f16, padded to 32 columns (cols 24..31 = 0) for the final WMMA.
// ---------------------------------------------------------------------------
__global__ void k_pts(const float* __restrict__ coords,
                      const float* __restrict__ vpts,
                      const float* __restrict__ mask,
                      _Float16* __restrict__ att) {
  int i    = blockIdx.x * (blockDim.x >> 5) + (threadIdx.x >> 5);
  int lane = threadIdx.x & 31;
  if (i >= S_) return;
  float cx = coords[i * 3], cy = coords[i * 3 + 1], cz = coords[i * 3 + 2];
  float mi = mask[i];
  float accv[24];
  #pragma unroll
  for (int c = 0; c < 24; ++c) accv[c] = 0.f;
  float wsum = 0.f;
  for (int j = lane; j < S_; j += 32) {
    float dx = cx - coords[j * 3], dy = cy - coords[j * 3 + 1], dz = cz - coords[j * 3 + 2];
    float dist = sqrtf(dx * dx + dy * dy + dz * dz + 1e-12f);
    float wg = __expf(-dist * 0.1f) * mi * mask[j];
    wsum += wg;
    const float* vp = vpts + (size_t)j * 24;
    #pragma unroll
    for (int c = 0; c < 24; ++c) accv[c] += wg * vp[c];
  }
  #pragma unroll
  for (int mm = 16; mm >= 1; mm >>= 1) {
    wsum += __shfl_xor(wsum, mm, 32);
    #pragma unroll
    for (int c = 0; c < 24; ++c) accv[c] += __shfl_xor(accv[c], mm, 32);
  }
  if (lane == 0) {
    float invw = 1.0f / (wsum + 1e-8f);
    #pragma unroll
    for (int c = 0; c < 24; ++c) att[(size_t)i * 32 + c] = (_Float16)(accv[c] * invw);
    #pragma unroll
    for (int c = 24; c < 32; ++c) att[(size_t)i * 32 + c] = (_Float16)0.0f;
  }
}

// ---------------------------------------------------------------------------
// Kernel 7: final combine as WMMA GEMMs.
//   out[i,c] = s[i,c] + ws*(ao@Wo^T + bo)[i,c] + wp*(att@W_po^T + b_po)[i,c]
// One wave per 16x16 output tile: K=256 (8 WMMA) for the scalar branch,
// K=32 (1 WMMA, zero-padded) for the point branch.
// ---------------------------------------------------------------------------
__global__ void k_out_wmma(const float* __restrict__ s,
                           const _Float16* __restrict__ aof,
                           const _Float16* __restrict__ attf,
                           const float* __restrict__ Wo,
                           const float* __restrict__ bo,
                           const float* __restrict__ Wpo,
                           const float* __restrict__ bpo,
                           const float* __restrict__ wsc,
                           const float* __restrict__ wpt,
                           float* __restrict__ out) {
  int tiles_n = CS_ >> 4;   // 24
  int waveId  = blockIdx.x * (blockDim.x >> 5) + (threadIdx.x >> 5);
  int lane    = threadIdx.x & 31;
  int ti = waveId / tiles_n, tn = waveId % tiles_n;
  if (ti >= (S_ >> 4)) return;
  int i0 = ti << 4, n0 = tn << 4;
  int M  = i0 + (lane & 15);
  int N  = n0 + (lane & 15);
  int hi = (lane >> 4) & 1;

  // scalar branch: ao (16x256 f16) @ Wo^T
  v8f acc = {};
  for (int k0 = 0; k0 < CH_; k0 += 32) {
    v16h a = load_a_frag(aof + (size_t)M * CH_ + k0, lane);
    const float* wrow = Wo + (size_t)N * CH_ + k0 + 16 * hi; // 16 contiguous f32
    v16h bm;
    #pragma unroll
    for (int e = 0; e < 16; ++e) bm[e] = (_Float16)wrow[e];
    acc = __builtin_amdgcn_wmma_f32_16x16x32_f16(false, a, false, bm,
                                                 (short)0, acc, false, false);
  }
  // point branch: att (16x32 f16, cols>=24 are zero) @ W_po^T (padded)
  v8f accp = {};
  {
    v16h a = load_a_frag(attf + (size_t)M * 32, lane);
    v16h bm;
    #pragma unroll
    for (int e = 0; e < 16; ++e) {
      int kk = 16 * hi + e;
      bm[e] = (kk < 24) ? (_Float16)Wpo[(size_t)N * 24 + kk] : (_Float16)0.0f;
    }
    accp = __builtin_amdgcn_wmma_f32_16x16x32_f16(false, a, false, bm,
                                                  (short)0, accp, false, false);
  }

  float ws = wsc[0], wp = wpt[0];
  #pragma unroll
  for (int r = 0; r < 8; ++r) {
    int i = i0 + r + 8 * hi;
    int c = n0 + (lane & 15);
    out[(size_t)i * CS_ + c] = s[(size_t)i * CS_ + c]
                             + ws * (acc[r] + bo[c])
                             + wp * (accp[r] + bpo[c]);
  }
}

// ---------------------------------------------------------------------------
extern "C" void kernel_launch(void* const* d_in, const int* in_sizes, int n_in,
                              void* d_out, int out_size, void* d_ws, size_t ws_size,
                              hipStream_t stream) {
  (void)in_sizes; (void)n_in; (void)out_size; (void)ws_size;
  const float* s      = (const float*)d_in[0];
  const float* z      = (const float*)d_in[1];
  const float* rigids = (const float*)d_in[2];
  const float* mask   = (const float*)d_in[3];
  const float* ln_s_g = (const float*)d_in[4];
  const float* ln_s_b = (const float*)d_in[5];
  const float* ln_z_g = (const float*)d_in[6];
  const float* ln_z_b = (const float*)d_in[7];
  const float* Wq     = (const float*)d_in[8];
  const float* Wk     = (const float*)d_in[9];
  const float* Wv     = (const float*)d_in[10];
  const float* Wpair  = (const float*)d_in[11];
  const float* Wo     = (const float*)d_in[12];
  const float* bo     = (const float*)d_in[13];
  const float* Wvp    = (const float*)d_in[14];
  const float* bvp    = (const float*)d_in[15];
  const float* Wpo    = (const float*)d_in[16];
  const float* bpo    = (const float*)d_in[17];
  const float* wsc    = (const float*)d_in[18];
  const float* wpt    = (const float*)d_in[19];
  float* out = (float*)d_out;

  // workspace carve-up (~37 MB total; bias dominates)
  char* wsb = (char*)d_ws;
  size_t off = 0;
  auto carve = [&](size_t bytes) -> void* {
    void* p = wsb + off;
    off = (off + bytes + 255) & ~(size_t)255;
    return p;
  };
  _Float16* s_n   = (_Float16*)carve((size_t)S_ * CS_ * 2);
  _Float16* qf    = (_Float16*)carve((size_t)S_ * CH_ * 2);
  _Float16* kf    = (_Float16*)carve((size_t)S_ * CH_ * 2);
  _Float16* vf    = (_Float16*)carve((size_t)S_ * CH_ * 2);
  float*    pt    = (float*)carve((size_t)S_ * H_ * PV_ * 4);
  float*    crd   = (float*)carve((size_t)S_ * 3 * 4);
  float*    vpts  = (float*)carve((size_t)S_ * 24 * 4);
  float*    bias  = (float*)carve((size_t)H_ * S_ * S_ * 4);
  _Float16* aof   = (_Float16*)carve((size_t)S_ * CH_ * 2);
  _Float16* attf  = (_Float16*)carve((size_t)S_ * 32 * 2);

  // 1. LayerNorm(s) -> f16
  k_ln_s<<<S_ / 4, 128, 0, stream>>>(s, ln_s_g, ln_s_b, s_n);

  // 2. projections (WMMA): q, k, v (f16), pt (f32, +bvp)
  {
    int waves_qkv = (S_ / 16) * (CH_ / 16);   // 1024 waves
    k_proj<true><<<waves_qkv / 4, 128, 0, stream>>>(s_n, Wq, nullptr, (void*)qf, CH_);
    k_proj<true><<<waves_qkv / 4, 128, 0, stream>>>(s_n, Wk, nullptr, (void*)kf, CH_);
    k_proj<true><<<waves_qkv / 4, 128, 0, stream>>>(s_n, Wv, nullptr, (void*)vf, CH_);
    int waves_pt = (S_ / 16) * ((H_ * PV_) / 16); // 256 waves
    k_proj<false><<<waves_pt / 4, 128, 0, stream>>>(s_n, Wvp, bvp, (void*)pt, H_ * PV_);
  }

  // 3. coords + v_points
  k_vpoints<<<S_ / 256, 256, 0, stream>>>(rigids, pt, crd, vpts);

  // 4. pair bias (streams z once; bandwidth-dominant kernel)
  k_pairbias<<<(S_ * S_ / 16) / 4, 128, 0, stream>>>(z, ln_z_g, ln_z_b, Wpair, bias);

  // 5. flash attention (WMMA) -> ao f16
  k_attn<<<H_ * (S_ / 16) / 4, 128, 0, stream>>>(qf, kf, vf, bias, mask, aof);

  // 6. distance-weighted points -> att f16 (padded to 32)
  k_pts<<<S_ / 4, 128, 0, stream>>>(crd, vpts, mask, attf);

  // 7. final combine as WMMA: (S/16)*(CS/16) = 1536 waves -> 384 blocks
  k_out_wmma<<<((S_ / 16) * (CS_ / 16)) / 4, 128, 0, stream>>>(
      s, aof, attf, Wo, bo, Wpo, bpo, wsc, wpt, out);
}
